// NpuDeepseekV2MoE_88235808129208
// MI455X (gfx1250) — matrix operand order repaired and checked by reading, not compiled, not run on hardware
//
#include <hip/hip_runtime.h>
#include <hip/hip_bf16.h>
#include <math.h>

// Problem constants (from reference)
#define T_TOK   1024
#define HDIM    2048
#define NEXP    16
#define IDIM    1024
#define ISHARED 2048
#define TOPK    6
#define NGROUP  4
#define TOPKG   2
#define RSCALE  2.5f

// GEMM tiling
#define BM 128
#define BN 64
#define BK 32
#define LDA 40   // bf16 elements per LDS row (32 + 8 pad, keeps 16B alignment)
#define LDB 40

typedef __bf16 bf16_t;
typedef __attribute__((ext_vector_type(16))) __bf16 v16bf;
typedef __attribute__((ext_vector_type(8)))  __bf16 v8bf;
typedef __attribute__((ext_vector_type(8)))  float  v8f;
typedef __attribute__((ext_vector_type(4)))  unsigned int ui32x4;
typedef __attribute__((ext_vector_type(8)))  int i32x8;
typedef __attribute__((ext_vector_type(4)))  int i32x4;

// Build a 16-element bf16 fragment from two 16-byte LDS chunks (2x ds_load_b128).
static __device__ __forceinline__ v16bf make_frag(const bf16_t* lo, const bf16_t* hi) {
    union { v16bf v; v8bf h[2]; } u;
    u.h[0] = *(const v8bf*)lo;
    u.h[1] = *(const v8bf*)hi;
    return u.v;
}

// Issue a TDM 2D tile load: [tile_k x tile_m] bf16 tile from a [rows x stride]
// bf16 tensor at byte address ga, into LDS offset lds_off with row padding of
// 4 DWORDs after every 16 DWORDs (i.e. 32 bf16 + 8 pad = LDA layout).
// 6-arg builtin variant (clang-23 / therock-10.0 headers).
static __device__ __forceinline__ void tdm_load_tile_bf16(
        unsigned lds_off, unsigned long long ga,
        int kdim, int stride_elems) {
    ui32x4 g0;
    g0[0] = 1u;                                            // count=1, user mode
    g0[1] = lds_off;                                       // lds_addr
    g0[2] = (unsigned)(ga & 0xffffffffull);                // global_addr[31:0]
    g0[3] = (unsigned)((ga >> 32) & 0x01ffffffull)         // global_addr[56:32]
          | (2u << 30);                                    // type = 2 (image)
    i32x8 g1;
    g1[0] = (1 << 16)      // data_size = 1 -> 2 bytes (bf16)
          | (1 << 20)      // pad_enable
          | (3 << 22)      // pad_interval = 3 -> every 16 DWORDs (64B row)
          | (3 << 25);     // pad_amount   = 3 -> 4 DWORDs (16B = 8 bf16)
    g1[1] = (int)((unsigned)kdim << 16);                   // tensor_dim0[15:0]
    g1[2] = (int)(((unsigned)kdim >> 16)                   // tensor_dim0[31:16]
          | ((unsigned)T_TOK << 16));                      // tensor_dim1[15:0]
    g1[3] = (int)(((unsigned)T_TOK >> 16)                  // tensor_dim1[31:16]
          | ((unsigned)BK << 16));                         // tile_dim0 = 32
    g1[4] = BM;                                            // tile_dim1 = 128
    g1[5] = stride_elems;                                  // tensor_dim0_stride[31:0]
    g1[6] = 0;
    g1[7] = 0;
    i32x4 z4; z4[0] = 0; z4[1] = 0; z4[2] = 0; z4[3] = 0;
    i32x8 z8;
#pragma unroll
    for (int i = 0; i < 8; ++i) z8[i] = 0;
    __builtin_amdgcn_tensor_load_to_lds(g0, g1, z4, z4, z8, 0);
}

// ---------------------------------------------------------------------------
// Kernel 1: routing. One wave (32 lanes) per token.
// ---------------------------------------------------------------------------
__global__ __launch_bounds__(128)
void moe_route_kernel(const float* __restrict__ x,
                      const float* __restrict__ gate_w,
                      const float* __restrict__ bias,
                      float* __restrict__ combine) {
    const int lane = threadIdx.x & 31;
    const int wave = threadIdx.x >> 5;
    const int t = blockIdx.x * (blockDim.x >> 5) + wave;
    if (t >= T_TOK) return;

    const float4* xrow = (const float4*)(x + (size_t)t * HDIM);
    float4 xv[16];
#pragma unroll
    for (int i = 0; i < 16; ++i) xv[i] = xrow[lane + i * 32];

    float logit = 0.f;
    for (int e = 0; e < NEXP; ++e) {
        const float4* wrow = (const float4*)(gate_w + (size_t)e * HDIM);
        float acc = 0.f;
#pragma unroll
        for (int i = 0; i < 16; ++i) {
            float4 w = wrow[lane + i * 32];
            acc += xv[i].x * w.x + xv[i].y * w.y + xv[i].z * w.z + xv[i].w * w.w;
        }
#pragma unroll
        for (int off = 16; off >= 1; off >>= 1) acc += __shfl_xor(acc, off, 32);
        if (lane == e) logit = acc;
    }

    const bool act_lane = lane < NEXP;
    float b = bias[lane & (NEXP - 1)];
    float score = act_lane ? 1.f / (1.f + __expf(-logit)) : 0.f;
    float sc = act_lane ? score + b : -INFINITY;

    // Group score: sum of top-2 of the 4 experts in this lane's group.
    int gbase = lane & ~3;
    float a0 = __shfl(sc, gbase + 0, 32);
    float a1 = __shfl(sc, gbase + 1, 32);
    float a2 = __shfl(sc, gbase + 2, 32);
    float a3 = __shfl(sc, gbase + 3, 32);
    float m1 = fmaxf(fmaxf(a0, a1), fmaxf(a2, a3));
    float s2 = -INFINITY; bool rm = false;
    if (!rm && a0 == m1) rm = true; else s2 = fmaxf(s2, a0);
    if (!rm && a1 == m1) rm = true; else s2 = fmaxf(s2, a1);
    if (!rm && a2 == m1) rm = true; else s2 = fmaxf(s2, a2);
    if (!rm && a3 == m1) rm = true; else s2 = fmaxf(s2, a3);
    float gsum = m1 + s2;

    // Pick top-2 groups (lowest-index tie-break, like lax.top_k).
    float gs0 = __shfl(gsum, 0, 32), gs1 = __shfl(gsum, 4, 32);
    float gs2 = __shfl(gsum, 8, 32), gs3 = __shfl(gsum, 12, 32);
    float gsv[4] = { gs0, gs1, gs2, gs3 };
    int gi = (lane >> 2) & 3;
    int rank = 0;
#pragma unroll
    for (int j = 0; j < 4; ++j)
        rank += (gsv[j] > gsv[gi]) || (gsv[j] == gsv[gi] && j < gi);
    bool ingrp = act_lane && (rank < TOPKG);

    // Iterative top-6 over masked scores.
    float mval = ingrp ? sc : -INFINITY;
    bool selected = false;
    float wsum = 0.f;
#pragma unroll
    for (int it = 0; it < TOPK; ++it) {
        float m = mval;
#pragma unroll
        for (int off = 8; off >= 1; off >>= 1) m = fmaxf(m, __shfl_xor(m, off, 32));
        unsigned bal = __builtin_amdgcn_ballot_w32(act_lane && (mval == m) && (m > -INFINITY));
        int sel = __ffs((int)bal) - 1;
        float selscore = __shfl(score, sel, 32);
        wsum += selscore;
        if (lane == sel) { selected = true; mval = -INFINITY; }
    }

    if (act_lane) {
        float w = selected ? (score / wsum) * RSCALE : 0.f;
        combine[(size_t)t * NEXP + lane] = w;
    }
}

// ---------------------------------------------------------------------------
// Kernel 2: fp32 -> bf16 (vectorized, 8 elems/thread)
// ---------------------------------------------------------------------------
__global__ __launch_bounds__(256)
void cvt_bf16_kernel(const float* __restrict__ in, bf16_t* __restrict__ out, int n8) {
    int i = blockIdx.x * blockDim.x + threadIdx.x;
    if (i >= n8) return;
    const float4* p = (const float4*)in + (size_t)i * 2;
    float4 v0 = p[0], v1 = p[1];
    v8bf o;
    o[0] = (bf16_t)v0.x; o[1] = (bf16_t)v0.y; o[2] = (bf16_t)v0.z; o[3] = (bf16_t)v0.w;
    o[4] = (bf16_t)v1.x; o[5] = (bf16_t)v1.y; o[6] = (bf16_t)v1.z; o[7] = (bf16_t)v1.w;
    ((v8bf*)out)[i] = o;
}

// ---------------------------------------------------------------------------
// Kernel 3: gate_up GEMM + silu*up*combine fused epilogue.
// A tile staged by the Tensor Data Mover; B (fp32 weights) converted inline.
// ---------------------------------------------------------------------------
__global__ __launch_bounds__(256)
void moe_gateup_kernel(const bf16_t* __restrict__ xb,
                       const float* __restrict__ w_base, long w_stride,
                       const float* __restrict__ comb_base,
                       bf16_t* __restrict__ act_base, long act_stride,
                       int Ninter) {
    __shared__ bf16_t As[BM][LDA];
    __shared__ bf16_t Bg[BN][LDB];   // transposed: [n][k]
    __shared__ bf16_t Bu[BN][LDB];
    __shared__ float  cmb[BM];

    const int e  = blockIdx.z;
    const float* w   = w_base  + (long)e * w_stride;
    bf16_t*      act = act_base + (long)e * act_stride;
    const int n0 = blockIdx.x * BN;
    const int t0 = blockIdx.y * BM;
    const int tid  = threadIdx.x;
    const int lane = tid & 31;
    const int wave = tid >> 5;
    const int wm = wave & 3;
    const int wn = wave >> 2;
    const int gh = lane >> 4;
    const int lr = lane & 15;
    const int twoN = 2 * Ninter;
    const unsigned lds_as = (unsigned)(size_t)(void*)&As[0][0];

    if (tid < BM)
        cmb[tid] = comb_base ? comb_base[(size_t)(t0 + tid) * NEXP + e] : 1.0f;

    v8f accg[2][2], accu[2][2];
#pragma unroll
    for (int r = 0; r < 2; ++r)
#pragma unroll
        for (int c = 0; c < 2; ++c)
#pragma unroll
            for (int i = 0; i < 8; ++i) { accg[r][c][i] = 0.f; accu[r][c][i] = 0.f; }

    const int bk = tid >> 3;            // 0..31
    const int bn = (tid & 7) * 8;       // 0..56

    for (int k0 = 0; k0 < HDIM; k0 += BK) {
        // B tiles (gate and up) fp32 loads
        const float4* bgp = (const float4*)(w + (size_t)(k0 + bk) * twoN + (n0 + bn));
        const float4* bup = (const float4*)(w + (size_t)(k0 + bk) * twoN + (Ninter + n0 + bn));
        float4 bg0 = bgp[0], bg1 = bgp[1];
        float4 bu0 = bup[0], bu1 = bup[1];

        if (k0 + BK < HDIM) {   // global_prefetch_b8 of next K-slice
            __builtin_prefetch(w + (size_t)(k0 + BK + bk) * twoN + (n0 + bn), 0, 1);
            __builtin_prefetch(w + (size_t)(k0 + BK + bk) * twoN + (Ninter + n0 + bn), 0, 1);
        }

        __syncthreads();   // previous iteration finished consuming As/Bg/Bu

        // TDM: DMA the 128x32 bf16 A tile into padded LDS layout (one wave issues)
        if (wave == 0) {
            unsigned long long ga =
                (unsigned long long)(size_t)(xb + (size_t)t0 * HDIM + k0);
            tdm_load_tile_bf16(lds_as, ga, HDIM, HDIM);
        }

        // convert + transpose-store B while the TDM streams A
        float bgv[8] = { bg0.x, bg0.y, bg0.z, bg0.w, bg1.x, bg1.y, bg1.z, bg1.w };
        float buv[8] = { bu0.x, bu0.y, bu0.z, bu0.w, bu1.x, bu1.y, bu1.z, bu1.w };
#pragma unroll
        for (int j = 0; j < 8; ++j) {
            Bg[bn + j][bk] = (bf16_t)bgv[j];
            Bu[bn + j][bk] = (bf16_t)buv[j];
        }

        if (wave == 0) __builtin_amdgcn_s_wait_tensorcnt(0);
        __syncthreads();

        // fragments (layouts per CDNA5 ISA 7.12.2; two ds_load_b128 per frag)
        v16bf af[2], bgf[2], buf[2];
#pragma unroll
        for (int r = 0; r < 2; ++r) {
            const bf16_t* base = &As[wm * 32 + r * 16 + lr][0];
            af[r] = make_frag(base + gh * 8, base + 16 + gh * 8);
        }
#pragma unroll
        for (int c = 0; c < 2; ++c) {
            const bf16_t* gb = &Bg[wn * 32 + c * 16 + lr][0];
            bgf[c] = make_frag(gb + gh * 8, gb + 16 + gh * 8);
            const bf16_t* ub = &Bu[wn * 32 + c * 16 + lr][0];
            buf[c] = make_frag(ub + gh * 8, ub + 16 + gh * 8);
        }
#pragma unroll
        for (int r = 0; r < 2; ++r)
#pragma unroll
            for (int c = 0; c < 2; ++c) {
                accg[r][c] = __builtin_amdgcn_wmma_f32_16x16x32_bf16(
                    false, af[r], false, bgf[c], (short)0, accg[r][c], false, false);
                accu[r][c] = __builtin_amdgcn_wmma_f32_16x16x32_bf16(
                    false, af[r], false, buf[c], (short)0, accu[r][c], false, false);
            }
    }

    // epilogue: silu(g) * u * combine  -> bf16 activations
#pragma unroll
    for (int r = 0; r < 2; ++r)
#pragma unroll
        for (int c = 0; c < 2; ++c)
#pragma unroll
            for (int v = 0; v < 8; ++v) {
                int tl  = wm * 32 + r * 16 + gh * 8 + v;
                int col = n0 + wn * 32 + c * 16 + lr;
                float gv = accg[r][c][v];
                float uv = accu[r][c][v];
                float s  = gv / (1.f + __expf(-gv));
                float ao = s * uv * cmb[tl];
                act[(size_t)(t0 + tl) * Ninter + col] = (bf16_t)ao;
            }
}

// ---------------------------------------------------------------------------
// Kernel 4: fused down-projection over {shared, 16 experts}, accumulate f32.
// ---------------------------------------------------------------------------
__global__ __launch_bounds__(256)
void moe_down_kernel(const bf16_t* __restrict__ shared_act,
                     const float* __restrict__ shared_wd,
                     const bf16_t* __restrict__ expert_act,
                     const float* __restrict__ expert_wd,
                     float* __restrict__ out) {
    __shared__ bf16_t As[BM][LDA];
    __shared__ bf16_t Bs[BN][LDB];   // transposed [h][k]

    const int h0 = blockIdx.x * BN;
    const int t0 = blockIdx.y * BM;
    const int tid  = threadIdx.x;
    const int lane = tid & 31;
    const int wave = tid >> 5;
    const int wm = wave & 3;
    const int wn = wave >> 2;
    const int gh = lane >> 4;
    const int lr = lane & 15;
    const unsigned lds_as = (unsigned)(size_t)(void*)&As[0][0];

    v8f acc[2][2];
#pragma unroll
    for (int r = 0; r < 2; ++r)
#pragma unroll
        for (int c = 0; c < 2; ++c)
#pragma unroll
            for (int i = 0; i < 8; ++i) acc[r][c][i] = 0.f;

    const int bk = tid >> 3;
    const int bn = (tid & 7) * 8;

    for (int s = 0; s < 1 + NEXP; ++s) {
        const bf16_t* A  = (s == 0) ? shared_act
                                    : expert_act + (size_t)(s - 1) * T_TOK * IDIM;
        const float*  Bw = (s == 0) ? shared_wd
                                    : expert_wd + (size_t)(s - 1) * IDIM * HDIM;
        const int Kdim = (s == 0) ? ISHARED : IDIM;

        for (int k0 = 0; k0 < Kdim; k0 += BK) {
            const float4* bp = (const float4*)(Bw + (size_t)(k0 + bk) * HDIM + (h0 + bn));
            float4 b0 = bp[0], b1 = bp[1];

            if (k0 + BK < Kdim)
                __builtin_prefetch(Bw + (size_t)(k0 + BK + bk) * HDIM + (h0 + bn), 0, 1);

            __syncthreads();

            if (wave == 0) {
                unsigned long long ga =
                    (unsigned long long)(size_t)(A + (size_t)t0 * Kdim + k0);
                tdm_load_tile_bf16(lds_as, ga, Kdim, Kdim);
            }

            float bv[8] = { b0.x, b0.y, b0.z, b0.w, b1.x, b1.y, b1.z, b1.w };
#pragma unroll
            for (int j = 0; j < 8; ++j) Bs[bn + j][bk] = (bf16_t)bv[j];

            if (wave == 0) __builtin_amdgcn_s_wait_tensorcnt(0);
            __syncthreads();

            v16bf af[2], bf[2];
#pragma unroll
            for (int r = 0; r < 2; ++r) {
                const bf16_t* base = &As[wm * 32 + r * 16 + lr][0];
                af[r] = make_frag(base + gh * 8, base + 16 + gh * 8);
            }
#pragma unroll
            for (int c = 0; c < 2; ++c) {
                const bf16_t* bb = &Bs[wn * 32 + c * 16 + lr][0];
                bf[c] = make_frag(bb + gh * 8, bb + 16 + gh * 8);
            }
#pragma unroll
            for (int r = 0; r < 2; ++r)
#pragma unroll
                for (int c = 0; c < 2; ++c)
                    acc[r][c] = __builtin_amdgcn_wmma_f32_16x16x32_bf16(
                        false, af[r], false, bf[c], (short)0, acc[r][c], false, false);
        }
    }

    // exclusive tile -> plain f32 stores
#pragma unroll
    for (int r = 0; r < 2; ++r)
#pragma unroll
        for (int c = 0; c < 2; ++c)
#pragma unroll
            for (int v = 0; v < 8; ++v) {
                int tl  = wm * 32 + r * 16 + gh * 8 + v;
                int col = h0 + wn * 32 + c * 16 + lr;
                out[(size_t)(t0 + tl) * HDIM + col] = acc[r][c][v];
            }
}

// ---------------------------------------------------------------------------
extern "C" void kernel_launch(void* const* d_in, const int* in_sizes, int n_in,
                              void* d_out, int out_size, void* d_ws, size_t ws_size,
                              hipStream_t stream) {
    (void)in_sizes; (void)n_in; (void)out_size; (void)ws_size;

    const float* x          = (const float*)d_in[0];
    const float* gate_w     = (const float*)d_in[1];
    const float* bias       = (const float*)d_in[2];
    const float* shared_wgu = (const float*)d_in[3];
    const float* shared_wd  = (const float*)d_in[4];
    const float* exp_wgu    = (const float*)d_in[5];
    const float* exp_wd     = (const float*)d_in[6];
    float* out = (float*)d_out;

    // workspace layout (all 256B aligned)
    char* ws = (char*)d_ws;
    size_t off = 0;
    bf16_t* xb = (bf16_t*)(ws + off);      off += (size_t)T_TOK * HDIM * sizeof(bf16_t);
    float* combine = (float*)(ws + off);   off += (size_t)T_TOK * NEXP * sizeof(float);
    off = (off + 255) & ~(size_t)255;
    bf16_t* shared_act = (bf16_t*)(ws + off); off += (size_t)T_TOK * ISHARED * sizeof(bf16_t);
    bf16_t* expert_act = (bf16_t*)(ws + off);

    // 1) routing -> combine[T,16]
    moe_route_kernel<<<T_TOK / 4, 128, 0, stream>>>(x, gate_w, bias, combine);

    // 2) x -> bf16
    int n8 = T_TOK * HDIM / 8;
    cvt_bf16_kernel<<<n8 / 256, 256, 0, stream>>>(x, xb, n8);

    // 3) shared expert gate_up (+silu*mul) -> shared_act
    moe_gateup_kernel<<<dim3(ISHARED / BN, T_TOK / BM, 1), 256, 0, stream>>>(
        xb, shared_wgu, 0L, nullptr, shared_act, 0L, ISHARED);

    // 4) routed experts gate_up (+silu*mul*combine) -> expert_act
    moe_gateup_kernel<<<dim3(IDIM / BN, T_TOK / BM, NEXP), 256, 0, stream>>>(
        xb, exp_wgu, (long)HDIM * 2 * IDIM, combine, expert_act, (long)T_TOK * IDIM, IDIM);

    // 5) fused down-projection + accumulation over all sources -> out
    moe_down_kernel<<<dim3(HDIM / BN, T_TOK / BM, 1), 256, 0, stream>>>(
        shared_act, shared_wd, expert_act, exp_wd, out);
}
